// P_TO_V_Module_26259430048532
// MI455X (gfx1250) — matrix operand order repaired and checked by reading, not compiled, not run on hardware
//
#include <hip/hip_runtime.h>
#include <hip/hip_bf16.h>

typedef __attribute__((ext_vector_type(16))) _Float16 v16h;
typedef __attribute__((ext_vector_type(8)))  float    v8f;

#define NPTS    16384
#define NP_PER  8192
#define NVOX    4096
#define CFEAT   64

// ---------------------------------------------------------------------------
// Kernel 0: fold BN into weights -> f16 padded layout + f32 bias vectors
// ---------------------------------------------------------------------------
struct PrepArgs {
    const float* W[4];
    const float* gamma[4];
    const float* beta[4];
    const float* mean[4];
    const float* var[4];
    _Float16*    wOut[4];
    float*       bOut[4];
    int cout[4];
    int cin[4];
    int cinP[4];
};

__launch_bounds__(256)
__global__ void prep_kernel(PrepArgs pa) {
    const int L    = blockIdx.y;
    const int cout = pa.cout[L];
    const int cin  = pa.cin[L];
    const int cinP = pa.cinP[L];
    const int total = cout * cinP;
    for (int i = blockIdx.x * blockDim.x + threadIdx.x; i < total;
         i += gridDim.x * blockDim.x) {
        const int r = i / cinP;
        const int k = i - r * cinP;
        const float s = pa.gamma[L][r] * rsqrtf(pa.var[L][r] + 1e-5f);
        const float w = (k < cin) ? pa.W[L][r * cin + k] * s : 0.0f;
        pa.wOut[L][i] = (_Float16)w;
    }
    const int tid = blockIdx.x * blockDim.x + threadIdx.x;
    if (tid < cout) {
        const float s = pa.gamma[L][tid] * rsqrtf(pa.var[L][tid] + 1e-5f);
        pa.bOut[L][tid] = pa.beta[L][tid] - pa.mean[L][tid] * s;
    }
}

// ---------------------------------------------------------------------------
// Kernel 1: ball query, one wave32 per voxel, ordered first-k via ballot
// ---------------------------------------------------------------------------
__launch_bounds__(256)
__global__ void ballquery_kernel(const float* __restrict__ pc,
                                 const int*   __restrict__ vind,
                                 int* __restrict__ idx0,
                                 int* __restrict__ idx1) {
    const int wid  = blockIdx.x * 8 + (threadIdx.x >> 5);
    const int lane = threadIdx.x & 31;

    const int4 vi = ((const int4*)vind)[wid];
    const float vx = ((float)vi.w + 0.5f) * 0.4f + 0.0f;
    const float vy = ((float)vi.z + 0.5f) * 0.4f - 40.0f;
    const float vz = ((float)vi.y + 0.5f) * 1.0f - 3.0f;
    const int base = vi.x * NP_PER;

    int cnt0 = 0, cnt1 = 0, first0 = -1, first1 = -1;
    const unsigned below = (1u << lane) - 1u;

    for (int it = 0; it < NP_PER / 32; ++it) {
        if (cnt0 >= 16 && cnt1 >= 32) break;   // uniform
        const int p = base + it * 32 + lane;
        const float4 q = ((const float4*)pc)[p];
        const float dx = q.y - vx, dy = q.z - vy, dz = q.w - vz;
        const float d2 = dx * dx + dy * dy + dz * dz;
        const unsigned m1 = __builtin_amdgcn_ballot_w32(d2 < 0.64f);
        const unsigned m0 = __builtin_amdgcn_ballot_w32(d2 < 0.16f);
        if (m0 != 0u && cnt0 < 16) {           // uniform
            if (first0 < 0) first0 = base + it * 32 + __builtin_ctz(m0);
            if ((m0 >> lane) & 1u) {
                const int pos = cnt0 + __popc(m0 & below);
                if (pos < 16) idx0[wid * 16 + pos] = p;
            }
            cnt0 += __popc(m0); if (cnt0 > 16) cnt0 = 16;
        }
        if (m1 != 0u && cnt1 < 32) {           // uniform
            if (first1 < 0) first1 = base + it * 32 + __builtin_ctz(m1);
            if ((m1 >> lane) & 1u) {
                const int pos = cnt1 + __popc(m1 & below);
                if (pos < 32) idx1[wid * 32 + pos] = p;
            }
            cnt1 += __popc(m1); if (cnt1 > 32) cnt1 = 32;
        }
    }
    if (lane < 16 && lane >= cnt0) idx0[wid * 16 + lane] = first0;  // -1 == empty
    if (lane >= cnt1)              idx1[wid * 32 + lane] = first1;
}

// ---------------------------------------------------------------------------
// Kernel 2: gather + 2-layer MLP (WMMA f16) + max-over-samples
//   one wave per voxel, 4 waves per block
// ---------------------------------------------------------------------------
template <int NS, int COUT2>
__launch_bounds__(128)
__global__ void group_mlp_kernel(const float* __restrict__ pc,
                                 const float* __restrict__ feats,
                                 const int*   __restrict__ vind,
                                 const int*   __restrict__ idxArr,
                                 const _Float16* __restrict__ w1,  // [64][96]
                                 const _Float16* __restrict__ w2,  // [COUT2][64]
                                 const float* __restrict__ b1,
                                 const float* __restrict__ b2,
                                 float* __restrict__ out, int outOff) {
    __shared__ alignas(16) _Float16 sX[4][NS][96];
    __shared__ alignas(16) _Float16 sY[4][NS][64];

    const int wq   = threadIdx.x >> 5;
    const int lane = threadIdx.x & 31;
    const int v    = blockIdx.x * 4 + wq;

    const int4 vi = ((const int4*)vind)[v];
    const float vx = ((float)vi.w + 0.5f) * 0.4f + 0.0f;
    const float vy = ((float)vi.z + 0.5f) * 0.4f - 40.0f;
    const float vz = ((float)vi.y + 0.5f) * 1.0f - 3.0f;

    // ---- gather: build X[NS][96] f16 (cols 0..2 dxyz, 3..66 feats, 67.. pad)
    for (int s = 0; s < NS; ++s) {
        if (lane < 29) sX[wq][s][67 + lane] = (_Float16)0.0f;
        const int p = idxArr[v * NS + s];
        if (p >= 0) {
            if (lane < 3) {
                const float pcomp = pc[p * 4 + 1 + lane];
                const float vcomp = (lane == 0) ? vx : ((lane == 1) ? vy : vz);
                sX[wq][s][lane] = (_Float16)(pcomp - vcomp);
            }
            sX[wq][s][3  + lane] = (_Float16)feats[p * CFEAT + lane];
            sX[wq][s][35 + lane] = (_Float16)feats[p * CFEAT + 32 + lane];
        } else {  // empty ball -> zero group
            sX[wq][s][lane]      = (_Float16)0.0f;
            sX[wq][s][32 + lane] = (_Float16)0.0f;
            if (lane < 3) sX[wq][s][64 + lane] = (_Float16)0.0f;
        }
    }
    __syncthreads();

    const int n16 = lane & 15;
    const int hh  = lane >> 4;
    const int kb  = hh * 8;

    union V16 { v16h v; uint4 q[2]; };

    // ---- layer 1: Y = relu( X @ W1^T * bnS + bnB ), NS x 64, f16 -> sY
    for (int mt = 0; mt < NS / 16; ++mt) {
        V16 a[3];
        const int row = mt * 16 + n16;
        for (int kc = 0; kc < 3; ++kc) {
            a[kc].q[0] = *(const uint4*)&sX[wq][row][kc * 32 + kb];
            a[kc].q[1] = *(const uint4*)&sX[wq][row][kc * 32 + 16 + kb];
        }
        for (int nt = 0; nt < 4; ++nt) {
            const float bias = b1[nt * 16 + n16];
            v8f c = {bias, bias, bias, bias, bias, bias, bias, bias};
            for (int kc = 0; kc < 3; ++kc) {
                V16 bf;
                const uint4* wp =
                    (const uint4*)(w1 + (nt * 16 + n16) * 96 + kc * 32 + hh * 8);
                bf.q[0] = wp[0];
                bf.q[1] = wp[2];
                c = __builtin_amdgcn_wmma_f32_16x16x32_f16(
                        false, a[kc].v, false, bf.v, (short)0, c, false, false);
            }
            const int rb  = mt * 16 + hh * 8;
            const int col = nt * 16 + n16;
            for (int j = 0; j < 8; ++j) {
                const float y = c[j] > 0.0f ? c[j] : 0.0f;
                sY[wq][rb + j][col] = (_Float16)y;
            }
        }
    }
    __syncthreads();

    // ---- layer 2 + relu + max over samples
    for (int nt = 0; nt < COUT2 / 16; ++nt) {
        const float bias = b2[nt * 16 + n16];
        float maxv = -3.0e38f;
        for (int mt = 0; mt < NS / 16; ++mt) {
            v8f c = {bias, bias, bias, bias, bias, bias, bias, bias};
            const int row = mt * 16 + n16;
            for (int kc = 0; kc < 2; ++kc) {
                V16 a2, bf;
                a2.q[0] = *(const uint4*)&sY[wq][row][kc * 32 + kb];
                a2.q[1] = *(const uint4*)&sY[wq][row][kc * 32 + 16 + kb];
                const uint4* wp =
                    (const uint4*)(w2 + (nt * 16 + n16) * 64 + kc * 32 + hh * 8);
                bf.q[0] = wp[0];
                bf.q[1] = wp[2];
                c = __builtin_amdgcn_wmma_f32_16x16x32_f16(
                        false, a2.v, false, bf.v, (short)0, c, false, false);
            }
            for (int j = 0; j < 8; ++j) maxv = fmaxf(maxv, c[j]);
        }
        maxv = fmaxf(maxv, 0.0f);  // relu (max of relu == relu of max here)
        // cross-half max: SWAPX16 swizzle (xor_mask=0x10, and_mask=0x1f)
        const int o = __builtin_amdgcn_ds_swizzle(__float_as_int(maxv), 0x401f);
        maxv = fmaxf(maxv, __int_as_float(o));
        if (lane < 16) out[v * 160 + outOff + nt * 16 + n16] = maxv;
    }
}

// ---------------------------------------------------------------------------
extern "C" void kernel_launch(void* const* d_in, const int* in_sizes, int n_in,
                              void* d_out, int out_size, void* d_ws, size_t ws_size,
                              hipStream_t stream) {
    (void)in_sizes; (void)n_in; (void)out_size; (void)ws_size;
    const float* pc    = (const float*)d_in[0];
    const float* feats = (const float*)d_in[1];
    const int*   vind  = (const int*)d_in[2];

    char* ws = (char*)d_ws;
    int* idx0 = (int*)(ws + 0);          // 4096*16*4 = 262144
    int* idx1 = (int*)(ws + 262144);     // 4096*32*4 = 524288
    _Float16* wOut[4] = {
        (_Float16*)(ws + 786432),        // s0 l0: 64*96*2 = 12288
        (_Float16*)(ws + 798720),        // s0 l1: 64*64*2 =  8192
        (_Float16*)(ws + 806912),        // s1 l0: 64*96*2 = 12288
        (_Float16*)(ws + 819200)};       // s1 l1: 96*64*2 = 12288
    float* bOut[4] = {
        (float*)(ws + 831488),
        (float*)(ws + 831488 + 512),
        (float*)(ws + 831488 + 1024),
        (float*)(ws + 831488 + 1536)};

    PrepArgs pa;
    const int couts[4] = {64, 64, 64, 96};
    const int cins [4] = {67, 64, 67, 64};
    const int cinPs[4] = {96, 64, 96, 64};
    for (int L = 0; L < 4; ++L) {
        const int bi = 3 + L * 5;        // params flattened: W,gamma,beta,mean,var
        pa.W[L]     = (const float*)d_in[bi + 0];
        pa.gamma[L] = (const float*)d_in[bi + 1];
        pa.beta[L]  = (const float*)d_in[bi + 2];
        pa.mean[L]  = (const float*)d_in[bi + 3];
        pa.var[L]   = (const float*)d_in[bi + 4];
        pa.wOut[L]  = wOut[L];
        pa.bOut[L]  = bOut[L];
        pa.cout[L]  = couts[L];
        pa.cin[L]   = cins[L];
        pa.cinP[L]  = cinPs[L];
    }

    prep_kernel<<<dim3(6, 4), 256, 0, stream>>>(pa);
    ballquery_kernel<<<NVOX / 8, 256, 0, stream>>>(pc, vind, idx0, idx1);

    float* out = (float*)d_out;
    group_mlp_kernel<16, 64><<<NVOX / 4, 128, 0, stream>>>(
        pc, feats, vind, idx0, wOut[0], wOut[1], bOut[0], bOut[1], out, 0);
    group_mlp_kernel<32, 96><<<NVOX / 4, 128, 0, stream>>>(
        pc, feats, vind, idx1, wOut[2], wOut[3], bOut[2], bOut[3], out, 64);
}
